// GroupedQueryAttention_65343632441492
// MI455X (gfx1250) — matrix operand order repaired and checked by reading, not compiled, not run on hardware
//
#include <hip/hip_runtime.h>

// ---------------------------------------------------------------------------
// GQA forward for MI455X (gfx1250, wave32, WMMA 16x16x32 bf16,
// async global->LDS copies tracked by ASYNCcnt)
// b=2, n=2048, dim=1024, H=16, KVH=4, d=64
// ---------------------------------------------------------------------------

typedef __bf16 bf16_t;
typedef __attribute__((ext_vector_type(16))) __bf16 v16bf;
typedef __attribute__((ext_vector_type(8)))  float  v8f;

union FragBF { v16bf v; uint4 q[2]; };
union Acc8   { v8f v; float f[8]; };

__device__ __forceinline__ bf16_t f2bf(float f) {
  unsigned int x = __builtin_bit_cast(unsigned int, f);
  unsigned int r = x + 0x7FFFu + ((x >> 16) & 1u);           // RNE
  return __builtin_bit_cast(bf16_t, (unsigned short)(r >> 16));
}
__device__ __forceinline__ float bf2f(bf16_t h) {
  unsigned int x = ((unsigned int)__builtin_bit_cast(unsigned short, h)) << 16;
  return __builtin_bit_cast(float, x);
}

// Async global->LDS 16B-per-lane copy (CDNA5, ASYNCcnt).
// LDS destination offset = low 32 bits of the generic pointer (aperture maps
// flat addr[31:0] -> LDS byte offset).
__device__ __forceinline__ void async_load_b128(void* lds_ptr, const void* gptr) {
  unsigned lds_off = (unsigned)(unsigned long long)lds_ptr;
  asm volatile("global_load_async_to_lds_b128 %0, %1, off"
               :: "v"(lds_off), "v"(gptr)
               : "memory");
}
__device__ __forceinline__ void wait_async0() {
  asm volatile("s_wait_asynccnt 0x0" ::: "memory");
}

// ---------------------------------------------------------------------------
// f32 -> bf16 convert
// ---------------------------------------------------------------------------
__global__ __launch_bounds__(256) void cvt_f32_bf16(const float* __restrict__ src,
                                                    bf16_t* __restrict__ dst, int n) {
  int i = blockIdx.x * 256 + threadIdx.x;
  if (i < n) dst[i] = f2bf(src[i]);
}

// ---------------------------------------------------------------------------
// RoPE in-place on bf16 [rows][heads*64]; pos = row % seq
// ---------------------------------------------------------------------------
__global__ __launch_bounds__(256) void rope_kernel(bf16_t* __restrict__ X,
                                                   int rows, int heads, int ld, int seq) {
  int idx = blockIdx.x * 256 + threadIdx.x;
  int total = rows * heads * 32;
  if (idx >= total) return;
  int j    = idx & 31;
  int head = (idx >> 5) % heads;
  int row  = idx / (32 * heads);
  int pos  = row % seq;
  // inv_freq = 10000^(-2j/64)
  float inv = exp2f(-(float)(2 * j) * (1.0f / 64.0f) * 13.287712379549449f);
  float th  = (float)pos * inv;
  float c = cosf(th), s = sinf(th);
  bf16_t* p = X + (size_t)row * ld + head * 64;
  float x1 = bf2f(p[j]);
  float x2 = bf2f(p[j + 32]);
  p[j]      = f2bf(x1 * c - x2 * s);
  p[j + 32] = f2bf(x2 * c + x1 * s);
}

// ---------------------------------------------------------------------------
// Tiled WMMA GEMM: C[M,N] = A[M,K] * B[K,N]  (bf16 in, f32 accum)
// Block tile 128x128, 8 waves, wave tile 32x64, K-step 32.
// A tile: async global->LDS b128 (row-major, A-frag reads contiguous).
// B tile: manual transposed stores [n][k] (B-frag reads contiguous).
// ---------------------------------------------------------------------------
#define GP 48   // LDS pitch in bf16 elems (96B, 16B-aligned)

template <bool F32OUT>
__global__ __launch_bounds__(256) void gemm_kernel(const bf16_t* __restrict__ A,
                                                   const bf16_t* __restrict__ B,
                                                   void* __restrict__ C,
                                                   const float* __restrict__ bias,
                                                   int M, int Nn, int K) {
  __shared__ bf16_t As[128][GP];
  __shared__ bf16_t Bs[128][GP];

  const int tid  = threadIdx.x;
  const int lane = tid & 31;
  const int wave = tid >> 5;
  const int lrow = lane & 15;
  const int hi16 = (lane >> 4) & 1;
  const int wm   = wave & 3;    // 4 waves along M
  const int wn   = wave >> 2;   // 2 waves along N
  const int blockM = blockIdx.y * 128;
  const int blockN = blockIdx.x * 128;

  Acc8 acc[2][4];
  for (int i = 0; i < 2; ++i)
    for (int j = 0; j < 4; ++j)
      for (int r = 0; r < 8; ++r) acc[i][j].f[r] = 0.0f;

  const int nk = K / 32;
  for (int kb = 0; kb < nk; ++kb) {
    const int k0 = kb * 32;
    if (kb) __syncthreads();

    // --- A tile 128x32: async global->LDS, 16B per lane, 512 chunks ---
    for (int i = 0; i < 2; ++i) {
      int idx = tid + i * 256;           // 0..511
      int row = idx >> 2;                // 4x 16B chunks per 32-elem row
      int c8  = idx & 3;
      async_load_b128(&As[row][c8 * 8],
                      A + (size_t)(blockM + row) * K + k0 + c8 * 8);
    }
    // --- B tile 32x128: load u32, store transposed [n][k] ---
    for (int i = 0; i < 8; ++i) {
      int idx = tid + i * 256;           // 0..2047
      int kk  = idx >> 6;                // 64 u32 per row of 128 cols
      int c2  = idx & 63;
      const bf16_t* gp = B + (size_t)(k0 + kk) * Nn + blockN + c2 * 2;
      unsigned int v = *(const unsigned int*)gp;
      Bs[c2 * 2 + 0][kk] = __builtin_bit_cast(bf16_t, (unsigned short)(v & 0xffffu));
      Bs[c2 * 2 + 1][kk] = __builtin_bit_cast(bf16_t, (unsigned short)(v >> 16));
      if (kb + 1 < nk) __builtin_prefetch(gp + (size_t)32 * Nn, 0, 1);
    }
    wait_async0();
    __syncthreads();

    // --- fragments + 8 WMMAs per wave ---
    FragBF a[2], b4[4];
    const int koffA = hi16 ? 8 : 0;
    for (int sm = 0; sm < 2; ++sm) {
      const bf16_t* base = &As[wm * 32 + sm * 16 + lrow][0];
      a[sm].q[0] = *(const uint4*)(base + koffA);
      a[sm].q[1] = *(const uint4*)(base + 16 + koffA);
    }
    const int kbB = hi16 ? 8 : 0;
    for (int sn = 0; sn < 4; ++sn) {
      const bf16_t* base = &Bs[wn * 64 + sn * 16 + lrow][0];
      b4[sn].q[0] = *(const uint4*)(base + kbB);
      b4[sn].q[1] = *(const uint4*)(base + kbB + 16);
    }
    for (int sm = 0; sm < 2; ++sm)
      for (int sn = 0; sn < 4; ++sn)
        acc[sm][sn].v = __builtin_amdgcn_wmma_f32_16x16x32_bf16(
            false, a[sm].v, false, b4[sn].v, (short)0, acc[sm][sn].v, false, false);
  }

  // --- store ---
  const int mOff = hi16 ? 8 : 0;
  for (int sm = 0; sm < 2; ++sm)
    for (int sn = 0; sn < 4; ++sn) {
      int n     = blockN + wn * 64 + sn * 16 + lrow;
      int mBase = blockM + wm * 32 + sm * 16 + mOff;
      for (int r = 0; r < 8; ++r) {
        size_t off = (size_t)(mBase + r) * Nn + n;
        if constexpr (F32OUT) ((float*)C)[off]  = acc[sm][sn].f[r] + bias[n];
        else                  ((bf16_t*)C)[off] = f2bf(acc[sm][sn].f[r]);
      }
    }
}

// ---------------------------------------------------------------------------
// Flash attention: one block = 64 query rows of one (batch, head); 4 waves,
// each wave owns 16 rows. Bc = 64 keys per tile, online softmax in f32.
// K tile loaded with async global->LDS; V tile transposed manually.
// Q: [B*NS][1024] (col = h*64+d), KV: [B*NS][512] (K heads 0..3, V heads 4..7)
// O: [B*NS][1024] bf16
// ---------------------------------------------------------------------------
__global__ __launch_bounds__(128) void flash_kernel(const bf16_t* __restrict__ Q,
                                                    const bf16_t* __restrict__ KV,
                                                    bf16_t* __restrict__ O) {
  constexpr int NS = 2048;
  __shared__ bf16_t Ks[64][72];       // [key][d]   (B-frag for Q*K^T contiguous)
  __shared__ bf16_t Vt[64][72];       // [d][key]   (B-frag for P*V contiguous)
  __shared__ bf16_t Ps[4][16][72];    // per-wave P re-swizzle buffer

  const int tid  = threadIdx.x;
  const int lane = tid & 31;
  const int wave = tid >> 5;
  const int lrow = lane & 15;
  const int hi16 = (lane >> 4) & 1;

  const int h     = blockIdx.y;
  const int batch = blockIdx.z;
  const int kh    = h >> 2;                    // GQA group of 4
  const int qBase = blockIdx.x * 64 + wave * 16;

  // Q fragments (2x 16x32 bf16, A-layout), loaded straight from global
  FragBF qa[2];
  {
    const bf16_t* qp = Q + (size_t)(batch * NS + qBase + lrow) * 1024 + h * 64;
    const int koff = hi16 ? 8 : 0;
    for (int kf = 0; kf < 2; ++kf) {
      qa[kf].q[0] = *(const uint4*)(qp + kf * 32 + koff);
      qa[kf].q[1] = *(const uint4*)(qp + kf * 32 + 16 + koff);
    }
  }

  Acc8 o[4];
  float m_i[8], l_i[8];
  for (int t = 0; t < 4; ++t)
    for (int r = 0; r < 8; ++r) o[t].f[r] = 0.0f;
  for (int r = 0; r < 8; ++r) { m_i[r] = -1e30f; l_i[r] = 0.0f; }

  const float cs = 0.125f * 1.44269504088896340736f;   // scale * log2(e)
  const int kbB  = hi16 ? 8 : 0;
  const int koff = hi16 ? 8 : 0;
  const int mOff = hi16 ? 8 : 0;

  for (int kt = 0; kt < NS / 64; ++kt) {
    __syncthreads();
    const size_t keyRow0 = (size_t)batch * NS + (size_t)kt * 64;
    // K tile: async global->LDS (direct [key][d] copy), 512 16B chunks
    for (int i = 0; i < 4; ++i) {
      int idx = tid + i * 128;
      int key = idx >> 3;                // 8x 16B chunks per 64-elem row
      int c8  = idx & 7;
      async_load_b128(&Ks[key][c8 * 8],
                      KV + (keyRow0 + key) * 512 + kh * 64 + c8 * 8);
    }
    // V tile: transpose into [d][key]
    for (int i = 0; i < 16; ++i) {
      int idx = tid + i * 128;
      int key = idx >> 5;
      int c2  = idx & 31;
      unsigned int v =
          *(const unsigned int*)(KV + (keyRow0 + key) * 512 + (4 + kh) * 64 + c2 * 2);
      Vt[c2 * 2 + 0][key] = __builtin_bit_cast(bf16_t, (unsigned short)(v & 0xffffu));
      Vt[c2 * 2 + 1][key] = __builtin_bit_cast(bf16_t, (unsigned short)(v >> 16));
    }
    wait_async0();
    __syncthreads();

    // --- S = Q K^T : 4 accum tiles of 16 keys ---
    Acc8 s[4];
    for (int t = 0; t < 4; ++t)
      for (int r = 0; r < 8; ++r) s[t].f[r] = 0.0f;
    for (int t = 0; t < 4; ++t)
      for (int kf = 0; kf < 2; ++kf) {
        FragBF kb;
        const bf16_t* base = &Ks[t * 16 + lrow][kf * 32 + kbB];
        kb.q[0] = *(const uint4*)(base);
        kb.q[1] = *(const uint4*)(base + 16);
        s[t].v = __builtin_amdgcn_wmma_f32_16x16x32_bf16(
            false, qa[kf].v, false, kb.v, (short)0, s[t].v, false, false);
      }

    // --- online softmax (rows striped: lane half-groups own 8 rows each) ---
    float mnew[8], alpha[8], lsum[8];
    for (int r = 0; r < 8; ++r) {
      float x = fmaxf(fmaxf(s[0].f[r], s[1].f[r]), fmaxf(s[2].f[r], s[3].f[r])) * cs;
      x = fmaxf(x, __shfl_xor(x, 1, 32));
      x = fmaxf(x, __shfl_xor(x, 2, 32));
      x = fmaxf(x, __shfl_xor(x, 4, 32));
      x = fmaxf(x, __shfl_xor(x, 8, 32));
      mnew[r]  = fmaxf(m_i[r], x);
      alpha[r] = exp2f(m_i[r] - mnew[r]);
      lsum[r]  = 0.0f;
    }
    for (int t = 0; t < 4; ++t)
      for (int r = 0; r < 8; ++r) {
        float p = exp2f(s[t].f[r] * cs - mnew[r]);
        lsum[r] += p;
        Ps[wave][mOff + r][t * 16 + lrow] = f2bf(p);   // C-layout -> LDS
      }
    for (int r = 0; r < 8; ++r) {
      float x = lsum[r];
      x += __shfl_xor(x, 1, 32);
      x += __shfl_xor(x, 2, 32);
      x += __shfl_xor(x, 4, 32);
      x += __shfl_xor(x, 8, 32);
      l_i[r] = l_i[r] * alpha[r] + x;
      m_i[r] = mnew[r];
    }
    for (int t = 0; t < 4; ++t)
      for (int r = 0; r < 8; ++r) o[t].f[r] *= alpha[r];

    // --- O += P V : read P back as A-frags (wave-private, DS in-order) ---
    FragBF pa[2];
    for (int kf = 0; kf < 2; ++kf) {
      const bf16_t* base = &Ps[wave][lrow][kf * 32 + koff];
      pa[kf].q[0] = *(const uint4*)(base);
      pa[kf].q[1] = *(const uint4*)(base + 16);
    }
    for (int t = 0; t < 4; ++t)
      for (int kf = 0; kf < 2; ++kf) {
        FragBF vb;
        const bf16_t* base = &Vt[t * 16 + lrow][kf * 32 + kbB];
        vb.q[0] = *(const uint4*)(base);
        vb.q[1] = *(const uint4*)(base + 16);
        o[t].v = __builtin_amdgcn_wmma_f32_16x16x32_bf16(
            false, pa[kf].v, false, vb.v, (short)0, o[t].v, false, false);
      }
  }

  // --- epilogue: O /= l, store bf16 [b*n][h*64+d] ---
  for (int t = 0; t < 4; ++t)
    for (int r = 0; r < 8; ++r) {
      float val = o[t].f[r] / l_i[r];
      size_t row = (size_t)(batch * NS + qBase + mOff + r);
      O[row * 1024 + h * 64 + t * 16 + lrow] = f2bf(val);
    }
}

// ---------------------------------------------------------------------------
// Host-side orchestration
// ---------------------------------------------------------------------------
extern "C" void kernel_launch(void* const* d_in, const int* in_sizes, int n_in,
                              void* d_out, int out_size, void* d_ws, size_t ws_size,
                              hipStream_t stream) {
  const float* x   = (const float*)d_in[0];   // [2,2048,1024]
  const float* Wq  = (const float*)d_in[1];   // [1024,1024]
  const float* Wkv = (const float*)d_in[2];   // [1024,512]
  const float* Wo  = (const float*)d_in[3];   // [1024,1024]
  const float* bo  = (const float*)d_in[4];   // [1024]
  float* out = (float*)d_out;

  constexpr int M   = 4096;   // b*n
  constexpr int DIM = 1024;
  constexpr int NKV = 512;

  char* ws = (char*)d_ws;
  bf16_t* xbf   = (bf16_t*)(ws + 0);                 // 8 MB
  bf16_t* Wqbf  = (bf16_t*)(ws + (8u  << 20));       // 2 MB
  bf16_t* Wkvbf = (bf16_t*)(ws + (10u << 20));       // 1 MB
  bf16_t* Wobf  = (bf16_t*)(ws + (11u << 20));       // 2 MB  (ends ~13MB)
  bf16_t* Qbf   = (bf16_t*)(ws + (13u << 20));       // 8 MB
  bf16_t* KVbf  = (bf16_t*)(ws + (21u << 20));       // 4 MB
  bf16_t* Obf   = (bf16_t*)(ws + (25u << 20));       // 8 MB (ends 33MB)

  // 1) f32 -> bf16 converts
  cvt_f32_bf16<<<(M * DIM + 255) / 256, 256, 0, stream>>>(x, xbf, M * DIM);
  cvt_f32_bf16<<<(DIM * DIM + 255) / 256, 256, 0, stream>>>(Wq, Wqbf, DIM * DIM);
  cvt_f32_bf16<<<(DIM * NKV + 255) / 256, 256, 0, stream>>>(Wkv, Wkvbf, DIM * NKV);
  cvt_f32_bf16<<<(DIM * DIM + 255) / 256, 256, 0, stream>>>(Wo, Wobf, DIM * DIM);

  // 2) projections (WMMA GEMMs)
  gemm_kernel<false><<<dim3(DIM / 128, M / 128), 256, 0, stream>>>(
      xbf, Wqbf, Qbf, nullptr, M, DIM, DIM);
  gemm_kernel<false><<<dim3(NKV / 128, M / 128), 256, 0, stream>>>(
      xbf, Wkvbf, KVbf, nullptr, M, NKV, DIM);

  // 3) RoPE on Q (16 heads) and K part of KV (first 4 heads of ld=512)
  rope_kernel<<<(M * 16 * 32 + 255) / 256, 256, 0, stream>>>(Qbf, M, 16, 1024, 2048);
  rope_kernel<<<(M * 4 * 32 + 255) / 256, 256, 0, stream>>>(KVbf, M, 4, 512, 2048);

  // 4) flash attention: grid (n/64, heads, batch)
  flash_kernel<<<dim3(2048 / 64, 16, 2), 128, 0, stream>>>(Qbf, KVbf, Obf);

  // 5) output projection + bias -> f32
  gemm_kernel<true><<<dim3(DIM / 128, M / 128), 256, 0, stream>>>(
      Obf, Wobf, out, bo, M, DIM, DIM);
}
